// NeuralEncoderDecoder_24885040513647
// MI455X (gfx1250) — matrix operand order repaired and checked by reading, not compiled, not run on hardware
//
#include <hip/hip_runtime.h>

typedef __attribute__((ext_vector_type(16))) __bf16 v16bf;
typedef __attribute__((ext_vector_type(8)))  __bf16 v8bf;
typedef __attribute__((ext_vector_type(8)))  float  v8f;

// ---------------------------------------------------------------------------
// helpers
// ---------------------------------------------------------------------------
__device__ __forceinline__ v8f vzero8() {
  v8f z;
#pragma unroll
  for (int i = 0; i < 8; ++i) z[i] = 0.0f;
  return z;
}

__device__ __forceinline__ v16bf pack_bf16(float4 q0, float4 q1, float4 q2, float4 q3) {
  v16bf a;
  a[0]  = (__bf16)q0.x; a[1]  = (__bf16)q0.y; a[2]  = (__bf16)q0.z; a[3]  = (__bf16)q0.w;
  a[4]  = (__bf16)q1.x; a[5]  = (__bf16)q1.y; a[6]  = (__bf16)q1.z; a[7]  = (__bf16)q1.w;
  a[8]  = (__bf16)q2.x; a[9]  = (__bf16)q2.y; a[10] = (__bf16)q2.z; a[11] = (__bf16)q2.w;
  a[12] = (__bf16)q3.x; a[13] = (__bf16)q3.y; a[14] = (__bf16)q3.z; a[15] = (__bf16)q3.w;
  return a;
}

__device__ __forceinline__ v16bf catbf(v8bf b0, v8bf b1) {
  v16bf b;
#pragma unroll
  for (int i = 0; i < 8; ++i) { b[i] = b0[i]; b[8 + i] = b1[i]; }
  return b;
}

__device__ __forceinline__ v8f wmma_bf16(v16bf a, v16bf b, v8f c) {
  return __builtin_amdgcn_wmma_f32_16x16x32_bf16(false, a, false, b, (short)0, c, false, false);
}

__device__ __forceinline__ float4 relu4(float4 q) {
  q.x = fmaxf(q.x, 0.0f); q.y = fmaxf(q.y, 0.0f);
  q.z = fmaxf(q.z, 0.0f); q.w = fmaxf(q.w, 0.0f);
  return q;
}

__device__ __forceinline__ float4 ldA4g(const float* __restrict__ row, int k, int K, bool ok) {
  if (ok && k < K) return *(const float4*)(row + k);
  return make_float4(0.0f, 0.0f, 0.0f, 0.0f);
}

// ---------------------------------------------------------------------------
// f32 row-major (K x 128) -> bf16 in WMMA-B fragment order, zero-padded to Kpad
// elem(k,c): kc=k/32, kk=k%32, n=c/16, lane=(c%16)+16*((kk/8)%2), j=kk%8+8*(kk/16)
// idx = (((kc*8 + n)*32 + lane)*16 + j)
// ---------------------------------------------------------------------------
__global__ void cvt_swizzle_bf16(const float* __restrict__ in, __bf16* __restrict__ out,
                                 int K, int Kpad) {
  int t = blockIdx.x * blockDim.x + threadIdx.x;
  int total = Kpad * 128;
  if (t >= total) return;
  int k = t >> 7, c = t & 127;
  float v = (k < K) ? in[(size_t)k * 128 + c] : 0.0f;
  int kc = k >> 5, kk = k & 31;
  int n = c >> 4, lc = c & 15;
  int l = lc + ((kk & 8) << 1);
  int j = (kk & 7) + ((kk & 16) >> 1);
  size_t o = ((((size_t)kc * 8 + n) * 32 + l) << 4) + (size_t)j;
  out[o] = (__bf16)v;
}

// ---------------------------------------------------------------------------
// C[M x 128] = A[M x K] (f32, row-major, lda) @ B (bf16, WMMA-swizzled)
// 256 threads = 8 wave32; each wave owns a 16-row M-tile x full N=128 (8 tiles)
// Main K-loop unguarded (row clamped); single guarded epilogue chunk for K%32.
// ---------------------------------------------------------------------------
__global__ void gemm_bf16wmma(const float* __restrict__ A, int lda,
                              const v8bf* __restrict__ Bsw,
                              float* __restrict__ C, int M, int K) {
  const int tid  = threadIdx.x;
  const int wave = tid >> 5;
  const int lane = tid & 31;
  const int lcol = lane & 15;
  const int m0   = (blockIdx.x * 8 + wave) * 16;
  const int row  = m0 + lcol;
  const bool rowOK = row < M;
  const float* Arow = A + (size_t)(rowOK ? row : 0) * (size_t)lda;
  const int soff = (lane < 16) ? 0 : 8;

  v8f acc[8];
#pragma unroll
  for (int n = 0; n < 8; ++n) acc[n] = vzero8();

  const int nFull = K >> 5;
  for (int kc = 0; kc < nFull; ++kc) {
    const int k0 = kc << 5;
    // stream-prefetch A ~1KB ahead (speculative, fault-safe)
    __builtin_prefetch(Arow + k0 + 256, 0, 3);
    float4 q0 = *(const float4*)(Arow + k0 + soff);
    float4 q1 = *(const float4*)(Arow + k0 + soff + 4);
    float4 q2 = *(const float4*)(Arow + k0 + soff + 16);
    float4 q3 = *(const float4*)(Arow + k0 + soff + 20);
    v16bf a = pack_bf16(q0, q1, q2, q3);
    v8bf bf[16];
#pragma unroll
    for (int n = 0; n < 8; ++n) {
      size_t bb = (((size_t)kc * 8 + n) * 32 + lane) * 2;
      bf[2 * n]     = Bsw[bb];
      bf[2 * n + 1] = Bsw[bb + 1];
    }
#pragma unroll
    for (int n = 0; n < 8; ++n)
      acc[n] = wmma_bf16(a, catbf(bf[2 * n], bf[2 * n + 1]), acc[n]);
  }

  if (K & 31) {                       // epilogue chunk with per-lane K guards
    const int kc = nFull;
    const int k0 = kc << 5;
    float4 q0 = ldA4g(Arow, k0 + soff,      K, rowOK);
    float4 q1 = ldA4g(Arow, k0 + soff + 4,  K, rowOK);
    float4 q2 = ldA4g(Arow, k0 + soff + 16, K, rowOK);
    float4 q3 = ldA4g(Arow, k0 + soff + 20, K, rowOK);
    v16bf a = pack_bf16(q0, q1, q2, q3);
    v8bf bf[16];
#pragma unroll
    for (int n = 0; n < 8; ++n) {
      size_t bb = (((size_t)kc * 8 + n) * 32 + lane) * 2;
      bf[2 * n]     = Bsw[bb];
      bf[2 * n + 1] = Bsw[bb + 1];
    }
#pragma unroll
    for (int n = 0; n < 8; ++n)
      acc[n] = wmma_bf16(a, catbf(bf[2 * n], bf[2 * n + 1]), acc[n]);
  }

  const int rbase = m0 + ((lane >= 16) ? 8 : 0);
#pragma unroll
  for (int n = 0; n < 8; ++n)
#pragma unroll
    for (int v = 0; v < 8; ++v) {
      int r = rbase + v;
      if (r < M) C[(size_t)r * 128 + n * 16 + lcol] = acc[n][v];
    }
}

// ---------------------------------------------------------------------------
// Edge decoder: per 16-edge group (one wave):
//   h = relu(concat(zi,zj)) @ W2   (16x256 @ 256x128, bf16 WMMA)
//   logit = h . W3[0:128] + (zi*zj) . W3[128:256];  out = sigmoid(logit)
// W2 (swizzled bf16, 64KB) async-copied into LDS once, reused every iteration.
// ---------------------------------------------------------------------------
__global__ void edge_decoder(const float* __restrict__ Z,
                             const __bf16* __restrict__ W2sw,
                             const float* __restrict__ W3,
                             const int* __restrict__ TE,
                             const int* __restrict__ TF,
                             int E1, int ngroups,
                             float* __restrict__ out) {
  __shared__ __align__(16) __bf16 sW2[256 * 128];
  __shared__ float sW3[256];
  __shared__ float sRed[8][16];

  const int tid = threadIdx.x;
  // ---- async bulk copy: global -> LDS, tracked by ASYNCcnt (no VGPR staging)
  {
    const char*    gsrc  = (const char*)W2sw;
    const uint32_t lbase = (uint32_t)(uintptr_t)(&sW2[0]);  // low 32b of flat = LDS offset
#pragma unroll
    for (int i = 0; i < 16; ++i) {
      uint32_t off = (uint32_t)(tid + 256 * i) * 16u;
      uint64_t ga  = (uint64_t)(uintptr_t)(gsrc + off);
      uint32_t la  = lbase + off;
      asm volatile("global_load_async_to_lds_b128 %0, %1, off"
                   :: "v"(la), "v"(ga) : "memory");
    }
    sW3[tid] = W3[tid];
    asm volatile("s_wait_asynccnt 0x0" ::: "memory");
  }
  __syncthreads();

  const int wave = tid >> 5;
  const int lane = tid & 31;
  const int lcol = lane & 15;
  const int soff = (lane < 16) ? 0 : 8;
  const int waveGlobal = blockIdx.x * 8 + wave;
  const int wavesTotal = gridDim.x * 8;
  const int iters = (ngroups + wavesTotal - 1) / wavesTotal;
  const v8bf* bp = (const v8bf*)sW2;

  for (int it = 0; it < iters; ++it) {
    const int g = waveGlobal + it * wavesTotal;
    const bool active = g < ngroups;
    const int gg = active ? g : 0;
    const int e = gg * 16 + lcol;              // this lane's edge (row of the tile)
    int i0, j0;
    if (e < E1) { i0 = TE[2 * e]; j0 = TE[2 * e + 1]; }
    else        { int e2 = e - E1; i0 = TF[2 * e2]; j0 = TF[2 * e2 + 1]; }
    const float* zi = Z + (size_t)i0 * 128;
    const float* zj = Z + (size_t)j0 * 128;

    v8f acc[8];
#pragma unroll
    for (int n = 0; n < 8; ++n) acc[n] = vzero8();

#pragma unroll
    for (int kc = 0; kc < 8; ++kc) {
      const float* pz = (kc < 4) ? (zi + kc * 32) : (zj + (kc - 4) * 32);
      float4 q0 = relu4(*(const float4*)(pz + soff));
      float4 q1 = relu4(*(const float4*)(pz + soff + 4));
      float4 q2 = relu4(*(const float4*)(pz + soff + 16));
      float4 q3 = relu4(*(const float4*)(pz + soff + 20));
      v16bf a = pack_bf16(q0, q1, q2, q3);
      v8bf bf[16];
#pragma unroll
      for (int n = 0; n < 8; ++n) {
        int bb = ((kc * 8 + n) * 32 + lane) * 2;
        bf[2 * n]     = bp[bb];
        bf[2 * n + 1] = bp[bb + 1];
      }
#pragma unroll
      for (int n = 0; n < 8; ++n)
        acc[n] = wmma_bf16(a, catbf(bf[2 * n], bf[2 * n + 1]), acc[n]);
    }

    // z_mult . W3[128:256]: each half-wave covers 64 of 128 cols for its row
    const int c0 = (lane >> 4) * 64;
    float mp = 0.0f;
#pragma unroll
    for (int c = 0; c < 64; c += 4) {
      float4 qa = *(const float4*)(zi + c0 + c);
      float4 qb = *(const float4*)(zj + c0 + c);
      float4 qw = *(const float4*)(&sW3[128 + c0 + c]);
      mp = fmaf(qa.x * qb.x, qw.x, mp);
      mp = fmaf(qa.y * qb.y, qw.y, mp);
      mp = fmaf(qa.z * qb.z, qw.z, mp);
      mp = fmaf(qa.w * qb.w, qw.w, mp);
    }
    mp += __shfl_xor(mp, 16, 32);              // lanes 0..15 hold full z_mult dot for row=lcol

    // h . W3[0:128]: C layout => lane holds (row = v + 8*(lane>=16), col = lcol)
    float w3r[8];
#pragma unroll
    for (int n = 0; n < 8; ++n) w3r[n] = sW3[n * 16 + lcol];
    float p[8];
#pragma unroll
    for (int v = 0; v < 8; ++v) {
      p[v] = 0.0f;
#pragma unroll
      for (int n = 0; n < 8; ++n) p[v] = fmaf(acc[n][v], w3r[n], p[v]);
    }
#pragma unroll
    for (int off = 8; off >= 1; off >>= 1)
#pragma unroll
      for (int v = 0; v < 8; ++v) p[v] += __shfl_xor(p[v], off, 32);

    if (lane == 0) {
#pragma unroll
      for (int v = 0; v < 8; ++v) sRed[wave][v] = p[v];
    }
    if (lane == 16) {
#pragma unroll
      for (int v = 0; v < 8; ++v) sRed[wave][8 + v] = p[v];
    }
    __syncthreads();
    if (active && lane < 16) {
      float logit = sRed[wave][lane] + mp;
      out[(size_t)g * 16 + lane] = 1.0f / (1.0f + __expf(-logit));
    }
    __syncthreads();
  }
}

// ---------------------------------------------------------------------------
// launcher
// ---------------------------------------------------------------------------
extern "C" void kernel_launch(void* const* d_in, const int* in_sizes, int n_in,
                              void* d_out, int out_size, void* d_ws, size_t ws_size,
                              hipStream_t stream) {
  (void)n_in; (void)out_size; (void)ws_size;
  const float* x   = (const float*)d_in[0];
  const float* adj = (const float*)d_in[1];
  const float* w1  = (const float*)d_in[2];
  const float* w2  = (const float*)d_in[3];
  const float* w3  = (const float*)d_in[4];
  const int*   te  = (const int*)d_in[5];
  const int*   tf  = (const int*)d_in[6];
  float* out = (float*)d_out;

  const int OUT  = 128;
  const int IN   = in_sizes[2] / OUT;                 // 512
  const int Nn   = in_sizes[0] / IN;                  // 10000
  const int E1   = in_sizes[5] / 2;                   // 262144
  const int totE = (in_sizes[5] + in_sizes[6]) / 2;   // 524288

  const int KpadW1 = (IN + 31) & ~31;                 // 512
  const int KpadW2 = (2 * OUT + 31) & ~31;            // 256
  const int KpadXW = (Nn + 31) & ~31;                 // 10016

  // workspace carve-out (all 256B aligned)
  char* ws = (char*)d_ws;
  size_t off = 0;
  auto carve = [&](size_t bytes) {
    void* p = ws + off;
    off += bytes;
    off = (off + 255) & ~(size_t)255;
    return p;
  };
  __bf16* w1sw = (__bf16*)carve((size_t)KpadW1 * 128 * sizeof(__bf16));
  __bf16* w2sw = (__bf16*)carve((size_t)KpadW2 * 128 * sizeof(__bf16));
  __bf16* xwsw = (__bf16*)carve((size_t)KpadXW * 128 * sizeof(__bf16));
  float*  xwf  = (float*) carve((size_t)Nn * 128 * sizeof(float));
  float*  zf   = (float*) carve((size_t)Nn * 128 * sizeof(float));

  // 1) swizzle weight, weight_two to bf16 WMMA-B layout
  cvt_swizzle_bf16<<<(KpadW1 * 128 + 255) / 256, 256, 0, stream>>>(w1, w1sw, IN, KpadW1);
  cvt_swizzle_bf16<<<(KpadW2 * 128 + 255) / 256, 256, 0, stream>>>(w2, w2sw, 2 * OUT, KpadW2);
  // 2) xw = x @ weight  (f32 out)
  gemm_bf16wmma<<<(Nn + 127) / 128, 256, 0, stream>>>(x, IN, (const v8bf*)w1sw, xwf, Nn, IN);
  // 3) xw -> bf16 swizzled (zero-padded K)
  cvt_swizzle_bf16<<<(KpadXW * 128 + 255) / 256, 256, 0, stream>>>(xwf, xwsw, Nn, KpadXW);
  // 4) z = adj @ xw
  gemm_bf16wmma<<<(Nn + 127) / 128, 256, 0, stream>>>(adj, Nn, (const v8bf*)xwsw, zf, Nn, Nn);
  // 5) edge decoder
  const int ngroups = totE / 16;
  edge_decoder<<<512, 256, 0, stream>>>(zf, w2sw, w3, te, tf, E1, ngroups, out);
}
